// ScalableGNN_19155554140466
// MI455X (gfx1250) — matrix operand order repaired and checked by reading, not compiled, not run on hardware
//
#include <hip/hip_runtime.h>

// ---------------------------------------------------------------------------
// Fused 2-layer GraphSAGE for MI455X (gfx1250, wave32, WMMA).
//   layer: out = relu?( (segsum(x[src])/max(cnt,1)) @ W_l + b_l + x @ W_r )
// Activations live in bf16 (halves the random edge-gather traffic, feeds WMMA
// directly); aggregation in fp32 via L2-resident atomics. Each wave computes a
// 16 x OUTC row strip: A fragments loaded once, reused across OUTC/16 WMMAs
// against LDS-staged transposed bf16 weights.
// ---------------------------------------------------------------------------

typedef __attribute__((ext_vector_type(16))) __bf16 v16bf;
typedef __attribute__((ext_vector_type(8)))  __bf16 v8bf;
typedef __attribute__((ext_vector_type(4)))  __bf16 v4bf;
typedef __attribute__((ext_vector_type(8)))  float  v8f;

#define NNODES 100000
#define K_DIM  128      // feature dim entering each layer (128 for both)

// --------------------------- utility: zero fill ----------------------------
__global__ void gnn_zero_f32(float* __restrict__ p, long n) {
  long i = (long)blockIdx.x * blockDim.x + threadIdx.x;
  long stride = (long)gridDim.x * blockDim.x;
  for (; i < n; i += stride) p[i] = 0.0f;
}

// ----------------------- utility: fp32 -> bf16 copy ------------------------
__global__ void gnn_cvt_bf16(const float* __restrict__ in,
                             __bf16* __restrict__ out, long n) {
  long i = (long)blockIdx.x * blockDim.x + threadIdx.x;
  long stride = (long)gridDim.x * blockDim.x;
  for (; i < n; i += stride) out[i] = (__bf16)in[i];
}

// ------------------- sparse: gather + scatter-add (mean) -------------------
// One wave32 per edge. Lane l handles channels [4l, 4l+3]: an 8B bf16 load per
// lane -> the wave streams the whole 256B source row coalesced, converts to
// fp32, then 4 fp32 atomics into agg[dst] (51MB target, L2-resident).
// Edge indices are wave-uniform -> readfirstlane keeps base addresses scalar.
__global__ void gnn_scatter(const __bf16* __restrict__ feat,
                            const long long* __restrict__ src,
                            const long long* __restrict__ dst,
                            float* __restrict__ agg,
                            float* __restrict__ cnt,
                            int nedges, int do_count) {
  int edge = blockIdx.x * (blockDim.x >> 5) + (threadIdx.x >> 5);
  int lane = threadIdx.x & 31;
  if (edge >= nedges) return;

  // node ids < 100000 fit in 32 bits
  int s = __builtin_amdgcn_readfirstlane((int)src[edge]);
  int d = __builtin_amdgcn_readfirstlane((int)dst[edge]);

  const v4bf v = *((const v4bf*)(feat + (long)s * K_DIM) + lane);
  float* a = agg + (long)d * K_DIM + lane * 4;
  __hip_atomic_fetch_add(a + 0, (float)v[0], __ATOMIC_RELAXED, __HIP_MEMORY_SCOPE_AGENT);
  __hip_atomic_fetch_add(a + 1, (float)v[1], __ATOMIC_RELAXED, __HIP_MEMORY_SCOPE_AGENT);
  __hip_atomic_fetch_add(a + 2, (float)v[2], __ATOMIC_RELAXED, __HIP_MEMORY_SCOPE_AGENT);
  __hip_atomic_fetch_add(a + 3, (float)v[3], __ATOMIC_RELAXED, __HIP_MEMORY_SCOPE_AGENT);
  if (do_count && lane == 0)
    __hip_atomic_fetch_add(cnt + d, 1.0f, __ATOMIC_RELAXED, __HIP_MEMORY_SCOPE_AGENT);
}

// ------------- dense: out = mean@Wl + b + x@Wr (+relu), via WMMA -----------
// One wave per 16-row strip, computing all OUTC/16 column tiles so each A
// fragment is loaded once and reused. 256-thread block = 8 waves = 8 strips.
// A operand (16x32 bf16, ISA 7.12.2): lane l -> row (l&15),
//   k = (l>>4)*16 + {0..15} (contiguous, same as B; not the ISA table order).
// B operand (32x16 bf16): lane l -> col (l&15), k = (l>>4)*16 + {0..15};
//   served from LDS sW[mat][col][k] (bf16, pitch 136 -> 272B rows: 16B-aligned
//   and bank-skewed so 16 distinct columns hit distinct banks).
// C/D (16x16 f32): vgpr r -> row r + 8*(l>>4), col (l&15).
template <int OUTC, bool RELU, typename OutT>
__global__ void gnn_sage_gemm(const __bf16* __restrict__ xself,
                              const float* __restrict__ agg,
                              const float* __restrict__ cnt,
                              const float* __restrict__ Wl,
                              const float* __restrict__ bl,
                              const float* __restrict__ Wr,
                              OutT* __restrict__ out, int nRowTiles) {
  constexpr int PITCH = 136;                 // bf16 elems per LDS column-row
  constexpr int CT = OUTC / 16;              // column tiles per wave
  __shared__ __align__(16) __bf16 sW[2][OUTC][PITCH];

  // --- cooperative convert + transpose of both weight matrices into LDS ---
  {
    const float* Wsrc[2] = {Wl, Wr};
#pragma unroll
    for (int m = 0; m < 2; ++m) {
      const float* W = Wsrc[m];
      for (int idx = threadIdx.x; idx < K_DIM * OUTC; idx += 256) {
        int k = idx / OUTC;                  // OUTC is a power of two
        int c = idx & (OUTC - 1);
        sW[m][c][k] = (__bf16)W[idx];        // coalesced global read
      }
    }
  }
  __syncthreads();

  const int wg = blockIdx.x * 8 + (threadIdx.x >> 5);   // = row tile index
  if (wg >= nRowTiles) return;

  const int lane = threadIdx.x & 31;
  const int half = lane >> 4;   // 0 or 1
  const int l16  = lane & 15;

  const int arow = wg * 16 + l16;        // A row this lane feeds
  const int akb  = half * 16;            // A k-subblock base
  const int bkb  = half * 16;            // B k-subblock base

  // mean scale folded into the bf16 conversion of the aggregate operand
  const float inv = 1.0f / fmaxf(cnt[arow], 1.0f);

  v8f acc[CT];
#pragma unroll
  for (int ct = 0; ct < CT; ++ct) acc[ct] = (v8f){};

  // ---- pass 0: mean (fp32 agg * inv, cvt to bf16) @ Wl ----
#pragma unroll
  for (int kk = 0; kk < K_DIM; kk += 32) {
    const float* ap = agg + (long)arow * K_DIM + kk + akb;
    v16bf a;
#pragma unroll
    for (int i = 0; i < 8; ++i) a[i]     = (__bf16)(ap[i] * inv);
#pragma unroll
    for (int i = 0; i < 8; ++i) a[8 + i] = (__bf16)(ap[8 + i] * inv);
#pragma unroll
    for (int ct = 0; ct < CT; ++ct) {
      const v16bf b = *(const v16bf*)&sW[0][ct * 16 + l16][kk + bkb];
      acc[ct] = __builtin_amdgcn_wmma_f32_16x16x32_bf16(
          false, a, false, b, (short)0, acc[ct], false, false);
    }
  }

  // ---- pass 1: self (bf16, direct b128 loads) @ Wr ----
#pragma unroll
  for (int kk = 0; kk < K_DIM; kk += 32) {
    const __bf16* ap = xself + (long)arow * K_DIM + kk + akb;
    const v8bf lo = *(const v8bf*)(ap);        // k = akb + 0..7
    const v8bf hi = *(const v8bf*)(ap + 8);    // k = akb + 8..15
    const v16bf a = __builtin_shufflevector(lo, hi,
        0, 1, 2, 3, 4, 5, 6, 7, 8, 9, 10, 11, 12, 13, 14, 15);
#pragma unroll
    for (int ct = 0; ct < CT; ++ct) {
      const v16bf b = *(const v16bf*)&sW[1][ct * 16 + l16][kk + bkb];
      acc[ct] = __builtin_amdgcn_wmma_f32_16x16x32_bf16(
          false, a, false, b, (short)0, acc[ct], false, false);
    }
  }

  // ---- epilogue: bias (+relu), store 8 rows x CT cols per lane ----
#pragma unroll
  for (int ct = 0; ct < CT; ++ct) {
    const int ocol = ct * 16 + l16;
    const float bias = bl[ocol];
#pragma unroll
    for (int r = 0; r < 8; ++r) {
      const int orow = wg * 16 + half * 8 + r;
      float v = acc[ct][r] + bias;
      if (RELU) v = fmaxf(v, 0.0f);
      out[(long)orow * OUTC + ocol] = (OutT)v;
    }
  }
}

// ------------------------------- launcher ----------------------------------
extern "C" void kernel_launch(void* const* d_in, const int* in_sizes, int n_in,
                              void* d_out, int out_size, void* d_ws, size_t ws_size,
                              hipStream_t stream) {
  (void)n_in; (void)out_size; (void)ws_size;
  const float*     x   = (const float*)d_in[0];
  const long long* ei  = (const long long*)d_in[1];
  const float*     Wl1 = (const float*)d_in[2];
  const float*     bl1 = (const float*)d_in[3];
  const float*     Wr1 = (const float*)d_in[4];
  const float*     Wl2 = (const float*)d_in[5];
  const float*     bl2 = (const float*)d_in[6];
  const float*     Wr2 = (const float*)d_in[7];
  float*           out = (float*)d_out;

  const int nE = in_sizes[1] / 2;          // edge_index is (2, E)
  const long long* src = ei;
  const long long* dst = ei + nE;

  // workspace carve-up (all offsets 16B aligned)
  float*  agg = (float*)d_ws;                          // [N,128] fp32
  float*  cnt = agg + (size_t)NNODES * K_DIM;          // [N]     fp32
  __bf16* xb  = (__bf16*)(cnt + NNODES);               // [N,128] bf16
  __bf16* h   = xb + (size_t)NNODES * K_DIM;           // [N,128] bf16

  const int edgeBlocks = (nE + 7) / 8;     // 8 edges (waves) per 256-thr block
  const int rowTiles   = NNODES / 16;      // 6250
  const int gemmBlocks = (rowTiles + 7) / 8;

  // ---- prep: x -> bf16 ----
  gnn_cvt_bf16<<<2048, 256, 0, stream>>>(x, xb, (long)NNODES * K_DIM);

  // ---- layer 1 ----
  gnn_zero_f32<<<2048, 256, 0, stream>>>(agg, (long)NNODES * K_DIM + NNODES);
  gnn_scatter<<<edgeBlocks, 256, 0, stream>>>(xb, src, dst, agg, cnt, nE, 1);
  gnn_sage_gemm<128, true, __bf16><<<gemmBlocks, 256, 0, stream>>>(
      xb, agg, cnt, Wl1, bl1, Wr1, h, rowTiles);

  // ---- layer 2 ----
  gnn_zero_f32<<<2048, 256, 0, stream>>>(agg, (long)NNODES * K_DIM);
  gnn_scatter<<<edgeBlocks, 256, 0, stream>>>(h, src, dst, agg, cnt, nE, 0);
  gnn_sage_gemm<64, false, float><<<gemmBlocks, 256, 0, stream>>>(
      h, agg, cnt, Wl2, bl2, Wr2, out, rowTiles);
}
